// DeltaFlowLoss_56470230008553
// MI455X (gfx1250) — compile-verified, matrix operand order checked by reading, not run on hardware
//
#include <hip/hip_runtime.h>
#include <hip/hip_bf16.h>
#include <math.h>

// ---------------------------------------------------------------------------
// Workspace layout (floats):
//   [0..35]      bucket sums/counts:  sum(low,mid,high), cnt(low,mid,high),
//                then per meta cid: sum[3], cnt[3]  (6 + 5*6 = 36)
//   [36..291]    inst_cnt[256]
//   [292..547]   inst_sum_pl[256]
//   [548..803]   inst_sum_sp[256]
//   [804..2083]  inst_meta_counts[256*5]
// ---------------------------------------------------------------------------
#define KSEG 256
#define NMETA 5
#define OFF_BUCK 0
#define OFF_CNT  36
#define OFF_PL   292
#define OFF_SP   548
#define OFF_META 804
#define WS_FLOATS 2084

typedef __attribute__((ext_vector_type(2)))  float    v2f;
typedef __attribute__((ext_vector_type(8)))  float    v8f;
typedef __attribute__((ext_vector_type(16))) _Float16 v16h;

__device__ __forceinline__ int meta_of(int c) {
    // reference order: default 4; ==0 -> 0; vehicle -> 1; pedestrian -> 2; wheeled -> 3
    if (c == 0) return 0;
    if (c == 7 || c == 8 || c == 9 || c == 10 || c == 12 || c == 13) return 1;
    if (c == 2 || c == 3 || c == 4) return 2;
    if (c == 6 || c == 11) return 3;
    return 4;
}

__global__ void zero_ws_kernel(float* __restrict__ ws) {
    int i = blockIdx.x * blockDim.x + threadIdx.x;
    if (i < WS_FLOATS) ws[i] = 0.0f;
}

// Streaming pass: 32 bytes/point from HBM; all reductions land in LDS
// (ds_add_f32) and are flushed once per block with global_atomic_add_f32.
__global__ void point_pass_kernel(const float* __restrict__ est,
                                  const float* __restrict__ gt,
                                  const int*   __restrict__ cls,
                                  const int*   __restrict__ inst,
                                  float* __restrict__ ws, int N) {
    __shared__ float sBuck[36];
    __shared__ float sCnt[KSEG];
    __shared__ float sPl[KSEG];
    __shared__ float sSp[KSEG];
    __shared__ float sMeta[KSEG * NMETA];

    const int tid = threadIdx.x;
    for (int j = tid; j < 36; j += blockDim.x) sBuck[j] = 0.0f;
    for (int j = tid; j < KSEG; j += blockDim.x) { sCnt[j] = 0.0f; sPl[j] = 0.0f; sSp[j] = 0.0f; }
    for (int j = tid; j < KSEG * NMETA; j += blockDim.x) sMeta[j] = 0.0f;
    __syncthreads();

    const int stride = gridDim.x * blockDim.x;
    for (int i = blockIdx.x * blockDim.x + tid; i < N; i += stride) {
        if (i + stride < N) {
            __builtin_prefetch(est + 3 * (size_t)(i + stride), 0, 0);
            __builtin_prefetch(gt  + 3 * (size_t)(i + stride), 0, 0);
        }
        const size_t b3 = 3 * (size_t)i;
        const float ex = est[b3 + 0], ey = est[b3 + 1], ez = est[b3 + 2];
        const float gx = gt[b3 + 0],  gy = gt[b3 + 1],  gz = gt[b3 + 2];

        const bool m = isfinite(ex) && isfinite(ey) && isfinite(ez) &&
                       isfinite(gx) && isfinite(gy) && isfinite(gz);
        if (!m) continue;  // masked-out points contribute to nothing in the reference

        const float dx = ex - gx, dy = ey - gy, dz = ez - gz;
        const float pl = sqrtf(dx * dx + dy * dy + dz * dz);
        const float sp = sqrtf(gx * gx + gy * gy + gz * gz) * 10.0f;  // /0.1

        const int b = (sp < 0.4f) ? 0 : ((sp <= 1.0f) ? 1 : 2);
        const int mc = meta_of(cls[i]);
        const int k = inst[i] & (KSEG - 1);

        atomicAdd(&sBuck[b], pl);
        atomicAdd(&sBuck[3 + b], 1.0f);
        atomicAdd(&sBuck[6 + mc * 6 + b], pl);
        atomicAdd(&sBuck[6 + mc * 6 + 3 + b], 1.0f);
        atomicAdd(&sCnt[k], 1.0f);
        atomicAdd(&sPl[k], pl);
        atomicAdd(&sSp[k], sp);
        atomicAdd(&sMeta[k * NMETA + mc], 1.0f);
    }
    __syncthreads();

    for (int j = tid; j < 36; j += blockDim.x)
        if (sBuck[j] != 0.0f) atomicAdd(&ws[OFF_BUCK + j], sBuck[j]);
    for (int j = tid; j < KSEG; j += blockDim.x) {
        if (sCnt[j] != 0.0f) atomicAdd(&ws[OFF_CNT + j], sCnt[j]);
        if (sPl[j]  != 0.0f) atomicAdd(&ws[OFF_PL  + j], sPl[j]);
        if (sSp[j]  != 0.0f) atomicAdd(&ws[OFF_SP  + j], sSp[j]);
    }
    for (int j = tid; j < KSEG * NMETA; j += blockDim.x)
        if (sMeta[j] != 0.0f) atomicAdd(&ws[OFF_META + j], sMeta[j]);
}

// Single-workgroup finalize. Thread k handles instance k; the 256 contrib /
// valid values are reduced cross-lane with chained V_WMMA_F32_16X16X4_F32
// (B = ones => D[m,n] = sum_k A[m,k] + C[m,n]; pure f32, no precision loss).
__global__ void finalize_kernel(const float* __restrict__ ws, float* __restrict__ out) {
    __shared__ float cArr[KSEG];
    __shared__ float vArr[KSEG];
    __shared__ float rowC[32];
    __shared__ float rowV[32];

    const float W[NMETA] = {0.1f, 1.0f, 2.0f, 2.5f, 1.5f};
    const int k = threadIdx.x;

    // per-instance statistics
    const float cnt  = ws[OFF_CNT + k];
    const float safe = fmaxf(cnt, 1.0f);
    const float sp_mean = ws[OFF_SP + k] / safe;
    float err = ws[OFF_PL + k] / safe;
    if (!isfinite(err)) err = 0.0f;  // nan_to_num

    int   mode = 0;
    float best = ws[OFF_META + k * NMETA];
    for (int c = 1; c < NMETA; ++c) {
        const float v = ws[OFF_META + k * NMETA + c];
        if (v > best) { best = v; mode = c; }  // first max on ties (argmax semantics)
    }
    const bool valid = (k > 0) && (cnt > 0.0f) && (sp_mean > 0.4f);
    cArr[k] = valid ? err * expf(err) * W[mode] : 0.0f;
    vArr[k] = valid ? 1.0f : 0.0f;
    __syncthreads();

    if (k < 32) {  // wave 0 only: EXEC all-ones as WMMA requires
        const int L = k;
        v8f accC = {};
        v8f accV = {};
#if defined(__gfx1250__) && __has_builtin(__builtin_amdgcn_wmma_f32_16x16x4_f32)
        // A 16x4 f32 layout: VGPR0 lane L<16 -> (M=L,K=0), lane L>=16 -> (M=L-16,K=2)
        //                    VGPR1 lane L<16 -> (M=L,K=1), lane L>=16 -> (M=L-16,K=3)
        // => per-lane pair (a[0],a[1]) = data[base], data[base+16], base = L<16 ? L : L+16
        const int base = (L < 16) ? L : (L + 16);
        v2f bones; bones[0] = 1.0f; bones[1] = 1.0f;
        for (int ch = 0; ch < 4; ++ch) {
            v2f a;
            a[0] = cArr[ch * 64 + base];
            a[1] = cArr[ch * 64 + base + 16];
            accC = __builtin_amdgcn_wmma_f32_16x16x4_f32(false, a, false, bones,
                                                         (short)0, accC, false, false);
            v2f av;
            av[0] = vArr[ch * 64 + base];
            av[1] = vArr[ch * 64 + base + 16];
            accV = __builtin_amdgcn_wmma_f32_16x16x4_f32(false, av, false, bones,
                                                         (short)0, accV, false, false);
        }
#else
        // Fallback: one f16 WMMA per array. Pack the 256 values into rows 0..7
        // of a 16x32 f16 A; lane L<16 holds (M=L, K=0..7 and 16..23),
        // lane L>=16 holds (M=L-16, K=8..15 and 24..31).
        v16h aC = {};
        v16h aV = {};
        v16h bO;
        for (int t = 0; t < 16; ++t) bO[t] = (_Float16)1.0f;
        const int row   = (L < 16) ? L : (L - 16);
        const int kbase = (L < 16) ? 0 : 8;
        if (row < 8) {
            for (int t = 0; t < 8; ++t) {
                aC[t]     = (_Float16)cArr[row * 32 + kbase + t];
                aC[8 + t] = (_Float16)cArr[row * 32 + 16 + kbase + t];
                aV[t]     = (_Float16)vArr[row * 32 + kbase + t];
                aV[8 + t] = (_Float16)vArr[row * 32 + 16 + kbase + t];
            }
        }
        accC = __builtin_amdgcn_wmma_f32_16x16x32_f16(false, aC, false, bO,
                                                      (short)0, accC, false, false);
        accV = __builtin_amdgcn_wmma_f32_16x16x32_f16(false, aV, false, bO,
                                                      (short)0, accV, false, false);
#endif
        // D layout: lane 0 VGPR j = rowsum(m=j), lane 16 VGPR j = rowsum(m=8+j)
        float rc = 0.0f, rv = 0.0f;
        for (int j = 0; j < 8; ++j) { rc += accC[j]; rv += accV[j]; }
        rowC[L] = rc;
        rowV[L] = rv;
    }
    __syncthreads();

    if (k == 0) {
        const float contribSum = rowC[0] + rowC[16];
        const float nvalid     = rowV[0] + rowV[16];

        float base_loss = 0.0f;
        for (int b = 0; b < 3; ++b) {
            const float c = ws[OFF_BUCK + 3 + b];
            base_loss += (c > 0.0f) ? (ws[OFF_BUCK + b] / c) : 0.0f;
        }
        const float BW[3] = {0.1f, 0.4f, 0.5f};
        float class_loss = 0.0f;
        for (int cid = 0; cid < NMETA; ++cid) {
            float acc = 0.0f;
            for (int b = 0; b < 3; ++b) {
                const float c = ws[OFF_BUCK + 6 + cid * 6 + 3 + b];
                acc += BW[b] * ((c > 0.0f) ? (ws[OFF_BUCK + 6 + cid * 6 + b] / c) : 0.0f);
            }
            class_loss += W[cid] * acc;
        }
        const float instance_loss = (nvalid > 0.0f) ? (contribSum / nvalid) : 0.0f;
        out[0] = base_loss + class_loss + instance_loss;
    }
}

extern "C" void kernel_launch(void* const* d_in, const int* in_sizes, int n_in,
                              void* d_out, int out_size, void* d_ws, size_t ws_size,
                              hipStream_t stream) {
    const float* est  = (const float*)d_in[0];
    const float* gt   = (const float*)d_in[1];
    const int*   cls  = (const int*)d_in[2];
    const int*   inst = (const int*)d_in[3];
    float* ws  = (float*)d_ws;
    float* out = (float*)d_out;
    const int N = in_sizes[0] / 3;

    zero_ws_kernel<<<(WS_FLOATS + 255) / 256, 256, 0, stream>>>(ws);

    int blocks = (N + 255) / 256;
    if (blocks > 512) blocks = 512;  // grid-stride; amortizes flush atomics
    if (blocks < 1) blocks = 1;
    point_pass_kernel<<<blocks, 256, 0, stream>>>(est, gt, cls, inst, ws, N);

    finalize_kernel<<<1, 256, 0, stream>>>(ws, out);
    (void)n_in; (void)out_size; (void)ws_size;
}
